// QuantizedLinear_4561255268710
// MI455X (gfx1250) — matrix operand order repaired and checked by reading, not compiled, not run on hardware
//
#include <hip/hip_runtime.h>
#include <cstdint>

// ---- types for CDNA5 WMMA ----
typedef __attribute__((ext_vector_type(16))) __bf16 v16bf;
typedef __attribute__((ext_vector_type(8)))  __bf16 v8bf;
typedef __attribute__((ext_vector_type(4)))  __bf16 v4bf;
typedef __attribute__((ext_vector_type(8)))  float  v8f;
typedef __attribute__((ext_vector_type(4)))  float  v4f;
typedef __attribute__((ext_vector_type(4)))  int    v4i;

#define D_IN   4096
#define D_OUT  11008
#define M_TOT  8192          // B*S = 4*2048
#define BM     128
#define BN     128
#define BK     64
#define LDK    (BK + 8)      // padded LDS row (bf16) -> 144B stride, conflict-free & 16B aligned
#define KTILES (D_IN / BK)   // 64

// ---------------- async helpers (CDNA5 gfx1250) ----------------
__device__ __forceinline__ void async_b128_to_lds(uint32_t lds_off, const void* gaddr) {
    // GLOBAL_LOAD_ASYNC_TO_LDS_B128: per-lane 16B memory->LDS, tracked by ASYNCcnt
    asm volatile("global_load_async_to_lds_b128 %0, %1, off"
                 :: "v"(lds_off), "v"(gaddr) : "memory");
}

__device__ __forceinline__ void wait_asynccnt0() {
#if defined(__has_builtin)
#if __has_builtin(__builtin_amdgcn_s_wait_asynccnt)
    __builtin_amdgcn_s_wait_asynccnt(0);
    return;
#endif
#endif
    asm volatile("s_wait_asynccnt 0x0" ::: "memory");
}

// ---------------- shared WMMA core ----------------
__device__ __forceinline__ void do_ksteps(__bf16 (*As)[LDK], __bf16 (*Bs)[LDK],
                                          int lane, int wm, int wn, v8f (&acc)[4][2])
{
    const int mrow0 = wm * 64 + (lane & 15);
    const int ncol0 = wn * 32 + (lane & 15);
    const int half  = lane >> 4;
#pragma unroll
    for (int ks = 0; ks < BK; ks += 32) {
        const int kb = ks + (half << 3);
        v16bf afrag[4];
#pragma unroll
        for (int mt = 0; mt < 4; ++mt) {
            const v8bf lo = *(const v8bf*)&As[mrow0 + mt * 16][kb];
            const v8bf hi = *(const v8bf*)&As[mrow0 + mt * 16][kb + 16];
            afrag[mt] = __builtin_shufflevector(lo, hi,
                0,1,2,3,4,5,6,7,8,9,10,11,12,13,14,15);
        }
        const int kg = ks + (half << 3);
        v16bf bfrag[2];
#pragma unroll
        for (int nt = 0; nt < 2; ++nt) {
            const v8bf lo = *(const v8bf*)&Bs[ncol0 + nt * 16][kg];
            const v8bf hi = *(const v8bf*)&Bs[ncol0 + nt * 16][kg + 16];
            bfrag[nt] = __builtin_shufflevector(lo, hi,
                0,1,2,3,4,5,6,7,8,9,10,11,12,13,14,15);
        }
#pragma unroll
        for (int mt = 0; mt < 4; ++mt)
#pragma unroll
            for (int nt = 0; nt < 2; ++nt)
                acc[mt][nt] = __builtin_amdgcn_wmma_f32_16x16x32_bf16(
                    false, afrag[mt], false, bfrag[nt],
                    (short)0, acc[mt][nt], false, false);
    }
}

__device__ __forceinline__ void epilogue(const float* __restrict__ bias,
                                         float* __restrict__ Out,
                                         int bm, int bn, int lane, int wm, int wn,
                                         v8f (&acc)[4][2])
{
    const int half    = lane >> 4;
    const int rowhalf = half << 3;
    const int ncol    = lane & 15;
#pragma unroll
    for (int mt = 0; mt < 4; ++mt) {
#pragma unroll
        for (int nt = 0; nt < 2; ++nt) {
            const int    gn  = bn * BN + wn * 32 + nt * 16 + ncol;
            const float  bv  = bias[gn];
            const size_t gm0 = (size_t)(bm * BM + wm * 64 + mt * 16 + rowhalf);
#pragma unroll
            for (int r = 0; r < 8; ++r)
                Out[(gm0 + r) * D_OUT + gn] = acc[mt][nt][r] + bv;
        }
    }
}

// ---------------- pre-conversion kernels ----------------
__global__ __launch_bounds__(256)
void convert_x_bf16(const float* __restrict__ X, __bf16* __restrict__ Xb) {
    const size_t i = ((size_t)blockIdx.x * 256 + threadIdx.x) * 4;
    const v4f xv = *(const v4f*)(X + i);
    v4bf xb;
    xb[0] = (__bf16)xv[0]; xb[1] = (__bf16)xv[1];
    xb[2] = (__bf16)xv[2]; xb[3] = (__bf16)xv[3];
    *(v4bf*)(Xb + i) = xb;
}

__global__ __launch_bounds__(256)
void dequant_w_bf16(const int* __restrict__ Wq, __bf16* __restrict__ Wb,
                    const float* __restrict__ scale_p, const float* __restrict__ zp_p) {
    const float scale = scale_p[0];
    const float zp    = zp_p[0];
    const size_t i = ((size_t)blockIdx.x * 256 + threadIdx.x) * 4;
    const v4i wv = *(const v4i*)(Wq + i);
    v4bf wb;
    wb[0] = (__bf16)(((float)wv[0] - zp) * scale);
    wb[1] = (__bf16)(((float)wv[1] - zp) * scale);
    wb[2] = (__bf16)(((float)wv[2] - zp) * scale);
    wb[3] = (__bf16)(((float)wv[3] - zp) * scale);
    *(v4bf*)(Wb + i) = wb;
}

// ---------------- main GEMM: async double-buffered staging ----------------
__global__ __launch_bounds__(256)
void qlinear_wmma_async(const __bf16* __restrict__ Xb, const __bf16* __restrict__ Wb,
                        const float* __restrict__ bias, float* __restrict__ Out)
{
    __shared__ __bf16 As[2][BM][LDK];
    __shared__ __bf16 Bs[2][BN][LDK];

    const int tid  = threadIdx.x;
    const int lane = tid & 31;
    const int wave = tid >> 5;
    const int wm   = wave >> 2;
    const int wn   = wave & 3;
    const int bn   = blockIdx.x;     // 0..85
    const int bm   = blockIdx.y;     // 0..63

    v8f acc[4][2];
    const v8f vzero = {0.f,0.f,0.f,0.f,0.f,0.f,0.f,0.f};
#pragma unroll
    for (int mt = 0; mt < 4; ++mt)
#pragma unroll
        for (int nt = 0; nt < 2; ++nt) acc[mt][nt] = vzero;

    // staging: tile = 128 rows x 8 chunks of 16B; 1024 chunks; 4 per thread per matrix
    const int crow = tid >> 3;       // 0..31, stepped by 32
    const int ccol = tid & 7;        // 16B chunk within row

    auto stage = [&](int buf, int kt) {
        const int kofs = kt * BK + ccol * 8;   // bf16 elements
#pragma unroll
        for (int i = 0; i < 4; ++i) {
            const int r = crow + i * 32;
            const __bf16* gx = Xb + (size_t)(bm * BM + r) * D_IN + kofs;
            async_b128_to_lds((uint32_t)(uintptr_t)&As[buf][r][ccol * 8], gx);
            const __bf16* gw = Wb + (size_t)(bn * BN + r) * D_IN + kofs;
            async_b128_to_lds((uint32_t)(uintptr_t)&Bs[buf][r][ccol * 8], gw);
        }
    };

    stage(0, 0);
    wait_asynccnt0();
    __syncthreads();

    for (int kt = 0; kt < KTILES; ++kt) {
        const int cur = kt & 1;
        if (kt + 1 < KTILES) stage(cur ^ 1, kt + 1);   // overlap: TDM-style async while WMMAs run
        do_ksteps(As[cur], Bs[cur], lane, wm, wn, acc);
        wait_asynccnt0();
        __syncthreads();
    }

    epilogue(bias, Out, bm, bn, lane, wm, wn, acc);
}

// ---------------- fallback: fused dequant (no workspace needed) ----------------
__global__ __launch_bounds__(256)
void qlinear_wmma_fused(const float* __restrict__ X, const int* __restrict__ Wq,
                        const float* __restrict__ scale_p, const float* __restrict__ zp_p,
                        const float* __restrict__ bias, float* __restrict__ Out)
{
    __shared__ __bf16 As[BM][LDK];
    __shared__ __bf16 Bs[BN][LDK];

    const int tid  = threadIdx.x;
    const int lane = tid & 31;
    const int wave = tid >> 5;
    const int wm   = wave >> 2;
    const int wn   = wave & 3;
    const int bn   = blockIdx.x;
    const int bm   = blockIdx.y;

    const float scale = scale_p[0];
    const float zp    = zp_p[0];

    v8f acc[4][2];
    const v8f vzero = {0.f,0.f,0.f,0.f,0.f,0.f,0.f,0.f};
#pragma unroll
    for (int mt = 0; mt < 4; ++mt)
#pragma unroll
        for (int nt = 0; nt < 2; ++nt) acc[mt][nt] = vzero;

    const int c4 = tid & 15;
    const int r0 = tid >> 4;

    for (int kt = 0; kt < KTILES; ++kt) {
        const int kbase = kt * BK;
#pragma unroll
        for (int i = 0; i < 8; ++i) {
            const int r = r0 + i * 16;
            const v4f xv = *(const v4f*)(X + (size_t)(bm * BM + r) * D_IN + kbase + c4 * 4);
            v4bf xb;
            xb[0] = (__bf16)xv[0]; xb[1] = (__bf16)xv[1];
            xb[2] = (__bf16)xv[2]; xb[3] = (__bf16)xv[3];
            *(v4bf*)&As[r][c4 * 4] = xb;

            const v4i wv = *(const v4i*)(Wq + (size_t)(bn * BN + r) * D_IN + kbase + c4 * 4);
            v4bf wb;
            wb[0] = (__bf16)(((float)wv[0] - zp) * scale);
            wb[1] = (__bf16)(((float)wv[1] - zp) * scale);
            wb[2] = (__bf16)(((float)wv[2] - zp) * scale);
            wb[3] = (__bf16)(((float)wv[3] - zp) * scale);
            *(v4bf*)&Bs[r][c4 * 4] = wb;
        }
        __syncthreads();

        if (kt + 1 < KTILES) {
            const int pr = tid & 127;
            __builtin_prefetch(X  + (size_t)(bm * BM + pr) * D_IN + kbase + BK, 0, 1);
            __builtin_prefetch(Wq + (size_t)(bn * BN + pr) * D_IN + kbase + BK, 0, 1);
        }

        do_ksteps(As, Bs, lane, wm, wn, acc);
        __syncthreads();
    }

    epilogue(bias, Out, bm, bn, lane, wm, wn, acc);
}

extern "C" void kernel_launch(void* const* d_in, const int* in_sizes, int n_in,
                              void* d_out, int out_size, void* d_ws, size_t ws_size,
                              hipStream_t stream) {
    (void)in_sizes; (void)n_in; (void)out_size;

    const float* x     = (const float*)d_in[0];
    const int*   wq    = (const int*)  d_in[1];
    const float* scale = (const float*)d_in[2];
    const float* zp    = (const float*)d_in[3];
    const float* bias  = (const float*)d_in[4];
    float*       out   = (float*)d_out;

    const size_t xb_bytes = (size_t)M_TOT * D_IN * sizeof(__bf16);   // 64 MiB
    const size_t wb_bytes = (size_t)D_OUT * D_IN * sizeof(__bf16);   // 86 MiB
    dim3 grid(D_OUT / BN, M_TOT / BM);                                // (86, 64)
    dim3 block(256);

    if (ws_size >= xb_bytes + wb_bytes) {
        // Pre-convert once: whole bf16 problem (150 MiB) then lives in the 192 MB L2.
        __bf16* xb = (__bf16*)d_ws;
        __bf16* wb = (__bf16*)((char*)d_ws + xb_bytes);

        convert_x_bf16<<<(unsigned)((size_t)M_TOT * D_IN / (4 * 256)), 256, 0, stream>>>(x, xb);
        dequant_w_bf16<<<(unsigned)((size_t)D_OUT * D_IN / (4 * 256)), 256, 0, stream>>>(wq, wb, scale, zp);
        qlinear_wmma_async<<<grid, block, 0, stream>>>(xb, wb, bias, out);
    } else {
        qlinear_wmma_fused<<<grid, block, 0, stream>>>(x, wq, scale, zp, bias, out);
    }
}
